// LinearAttentionBackbone_70068096467989
// MI455X (gfx1250) — matrix operand order repaired and checked by reading, not compiled, not run on hardware
//
#include <hip/hip_runtime.h>
#include <hip/hip_bf16.h>

typedef __attribute__((ext_vector_type(16))) __bf16 v16bf;
typedef __attribute__((ext_vector_type(8)))  __bf16 v8bf;
typedef __attribute__((ext_vector_type(4)))  __bf16 v4bf;
typedef __attribute__((ext_vector_type(8)))  float  v8f;

#define Dm   1024
#define NTOK 2048
#define NB   2
#define MROW 4096
#define VOC  50257

// ---------------- f32 -> bf16 bulk conversion (8 elems/thread) ------------
__global__ void cvt_f32_bf16(const float* __restrict__ in,
                             __bf16* __restrict__ out, long n) {
  long i = ((long)blockIdx.x * blockDim.x + threadIdx.x) * 8;
  if (i >= n) return;
  float4 x = *(const float4*)(in + i);
  float4 y = *(const float4*)(in + i + 4);
  v8bf o;
  o[0] = (__bf16)x.x; o[1] = (__bf16)x.y; o[2] = (__bf16)x.z; o[3] = (__bf16)x.w;
  o[4] = (__bf16)y.x; o[5] = (__bf16)y.y; o[6] = (__bf16)y.z; o[7] = (__bf16)y.w;
  *(v8bf*)(out + i) = o;
}

// ---------------- embedding gather: Xf/Xb[row] = emb[ids[row]] ------------
__global__ void gather_embed(const int* __restrict__ ids,
                             const float* __restrict__ emb,
                             float* __restrict__ Xf, __bf16* __restrict__ Xb) {
  int row = blockIdx.x;
  int id  = ids[row];
  float4 v = ((const float4*)(emb + (size_t)id * Dm))[threadIdx.x];
  ((float4*)(Xf + (size_t)row * Dm))[threadIdx.x] = v;
  v4bf o;
  o[0] = (__bf16)v.x; o[1] = (__bf16)v.y; o[2] = (__bf16)v.z; o[3] = (__bf16)v.w;
  *(v4bf*)(Xb + (size_t)row * Dm + threadIdx.x * 4) = o;
}

// ---- fragment helpers: uniform base + 32-bit BYTE offsets (saddr form) ---
struct AFrag { v8bf p0, q0, p1, q1; };

__device__ __forceinline__ v8bf ld8(const char* base, unsigned bo) {
  return *(const v8bf*)(base + bo);
}
__device__ __forceinline__ v16bf ld16(const char* base, unsigned bo) {
  return *(const v16bf*)(base + bo);
}

__device__ __forceinline__ AFrag loadA(const char* base, unsigned o0,
                                       unsigned o1, unsigned kb) {
  AFrag f;
  f.p0 = ld8(base, o0 + kb);
  f.q0 = ld8(base, o0 + kb + 16);   // +8 elements
  f.p1 = ld8(base, o1 + kb);
  f.q1 = ld8(base, o1 + kb + 16);
  return f;
}

__device__ __forceinline__ void loadB(const char* base, const unsigned* bo,
                                      unsigned kb, v16bf* b) {
#pragma unroll
  for (int t = 0; t < 4; t++) b[t] = ld16(base, bo[t] + kb);
}

__device__ __forceinline__ void mma8(const AFrag& f, const v16bf* b, v8f (*acc)[4]) {
  v16bf a0 = __builtin_shufflevector(f.p0, f.q0, 0,1,2,3,4,5,6,7,8,9,10,11,12,13,14,15);
  v16bf a1 = __builtin_shufflevector(f.p1, f.q1, 0,1,2,3,4,5,6,7,8,9,10,11,12,13,14,15);
#pragma unroll
  for (int t = 0; t < 4; t++) {
    acc[0][t] = __builtin_amdgcn_wmma_f32_16x16x32_bf16(
        false, a0, false, b[t], (short)0, acc[0][t], false, false);
    acc[1][t] = __builtin_amdgcn_wmma_f32_16x16x32_bf16(
        false, a1, false, b[t], (short)0, acc[1][t], false, false);
  }
}

// ---------------- generic WMMA GEMM: C = epilogue(A * Bt^T) ---------------
// A: M x K bf16 (K contiguous). Bt: N x K bf16 (K contiguous).
// C[m,n] = sum_k A[m,k]*Bt[n,k]. Epilogue: +bias[n], elu+1 (act),
// /zdiv[m]; write f32 (Cf) and/or bf16 (Cb).
// One wave -> 32M x 64N (8 accumulators). 2x-unrolled ping-pong pipeline;
// loop-carried byte offsets so loads lower to saddr + 32-bit voffset with
// immediate-offset companions. Kdim must be a multiple of 64.
__global__ void __launch_bounds__(256, 1)
gemm_wmma(const __bf16* __restrict__ Ag,
          const __bf16* __restrict__ Btg,
          const float* __restrict__ bias,
          const float* __restrict__ zdiv,
          float* __restrict__ Cfg,
          __bf16* __restrict__ Cbg,
          int Kdim, int Nlim, int ldc,
          long sA, long sB, long sC, long sZ, int act) {
  int batch = blockIdx.z;
  const char* A  = (const char*)(Ag  + (size_t)batch * sA);
  const char* Bt = (const char*)(Btg + (size_t)batch * sB);
  float*  Cf = Cfg ? Cfg + (size_t)batch * sC : nullptr;
  __bf16* Cb = Cbg ? Cbg + (size_t)batch * sC : nullptr;
  const float* zr = zdiv ? (zdiv + (size_t)batch * sZ) : nullptr;

  int wave = threadIdx.x >> 5;
  int lane = threadIdx.x & 31;
  int hi   = lane >> 4;
  int lo   = lane & 15;
  int M0   = blockIdx.x * 256 + wave * 32;
  int N0   = blockIdx.y * 64;

  // Byte offsets from the uniform A/Bt bases (loop-carried, +128 B/iter).
  // A layout: lanes 0-15 K k+0..15, lanes 16-31
  // K k+16..31. B: lane = column, lanes 0-15 K k..k+15,
  // lanes 16-31 K k+16..k+31 (32 contiguous bytes per lane).
  unsigned a0b = (unsigned)(((M0 + lo) * Kdim + hi * 16) * 2);
  unsigned a1b = a0b + (unsigned)(16 * Kdim * 2);
  unsigned bb[4];
#pragma unroll
  for (int t = 0; t < 4; t++) {
    int col = N0 + t * 16 + lo;
    int cc  = col < Nlim ? col : (Nlim - 1);   // clamp vocab tail (k-invariant)
    bb[t] = (unsigned)((cc * Kdim + hi * 16) * 2);
  }

  v8f acc[2][4] = {};

  // Ping-pong software pipeline (prefetch offsets are +64 B / +128 B).
  AFrag fa0 = loadA(A, a0b, a1b, 0);
  v16bf b0[4]; loadB(Bt, bb, 0, b0);
  AFrag fa1;
  v16bf b1[4];

  for (int k0 = 0; k0 + 64 < Kdim; k0 += 64) {
    fa1 = loadA(A, a0b, a1b, 64);
    loadB(Bt, bb, 64, b1);
    mma8(fa0, b0, acc);
    fa0 = loadA(A, a0b, a1b, 128);
    loadB(Bt, bb, 128, b0);
    mma8(fa1, b1, acc);
    a0b += 128; a1b += 128;
#pragma unroll
    for (int t = 0; t < 4; t++) bb[t] += 128;
  }
  // Peeled tail: last 64-wide chunk.
  fa1 = loadA(A, a0b, a1b, 64);
  loadB(Bt, bb, 64, b1);
  mma8(fa0, b0, acc);
  mma8(fa1, b1, acc);

  // C layout: lane<16 -> rows M0+0..7 (vgpr 0..7), lane>=16 -> rows M0+8..15.
#pragma unroll
  for (int s = 0; s < 2; s++) {
#pragma unroll
    for (int t = 0; t < 4; t++) {
      int col = N0 + t * 16 + lo;
      if (col < Nlim) {
        float bv = bias ? bias[col] : 0.0f;
#pragma unroll
        for (int v = 0; v < 8; v++) {
          int row   = M0 + s * 16 + hi * 8 + v;
          float val = acc[s][t][v] + bv;
          if (act) val = (val > 0.0f) ? (val + 1.0f) : __expf(val); // elu+1
          if (zr)  val = val / zr[row];
          if (Cf)  Cf[(size_t)row * ldc + col] = val;
          if (Cb)  Cb[(size_t)row * ldc + col] = (__bf16)val;
        }
      }
    }
  }
}

// -------- KVt[b][e][d] = sum_n V[b,n,e]*K[b,n,d]  (= (K^T V)^T) -----------
// Stored e-major bf16 so the context GEMM's B operand is K-contiguous.
__global__ void kv_wmma(const __bf16* __restrict__ Kmat,
                        const __bf16* __restrict__ Vmat,
                        __bf16* __restrict__ KVt) {
  int b = blockIdx.z;
  const __bf16* Kb = Kmat + (size_t)b * NTOK * Dm;
  const __bf16* Vb = Vmat + (size_t)b * NTOK * Dm;
  __bf16*       Ob = KVt  + (size_t)b * Dm * Dm;

  int wave = threadIdx.x >> 5;
  int lane = threadIdx.x & 31;
  int hi   = lane >> 4;
  int lo   = lane & 15;
  int M0   = blockIdx.x * 128 + wave * 16;   // e tile
  int N0   = blockIdx.y * 64;                // d tile group
  int eIdx = M0 + lo;

  v8f acc[4] = {};

  for (int k0 = 0; k0 < NTOK; k0 += 32) {
    v16bf a;   // A[m=e, k=n] = V[n,e] (strided gather; <2% of total FLOPs)
#pragma unroll
    for (int e = 0; e < 8; e++) {
      a[e]     = Vb[(size_t)(k0 + hi * 8 + e) * Dm + eIdx];
      a[e + 8] = Vb[(size_t)(k0 + 16 + hi * 8 + e) * Dm + eIdx];
    }
#pragma unroll
    for (int t = 0; t < 4; t++) {
      int d = N0 + t * 16 + lo;
      v16bf bb;  // B[k=n, col=d] = K[n,d]
#pragma unroll
      for (int e = 0; e < 16; e++)
        bb[e] = Kb[(size_t)(k0 + (e < 8 ? hi * 8 + e : 8 + hi * 8 + e)) * Dm + d];
      acc[t] = __builtin_amdgcn_wmma_f32_16x16x32_bf16(
          false, a, false, bb, (short)0, acc[t], false, false);
    }
  }
#pragma unroll
  for (int t = 0; t < 4; t++) {
    int d = N0 + t * 16 + lo;
#pragma unroll
    for (int v = 0; v < 8; v++)
      Ob[(size_t)(M0 + hi * 8 + v) * Dm + d] = (__bf16)acc[t][v];
  }
}

// -------- KsumEps[b,d] = sum_n K[b,n,d] + EPS (f32 accumulate) ------------
__global__ void ksum_kernel(const __bf16* __restrict__ Kmat,
                            float* __restrict__ KsumEps) {
  int idx = blockIdx.x * blockDim.x + threadIdx.x;   // b*Dm + d
  int b = idx >> 10, d = idx & (Dm - 1);
  const __bf16* kb = Kmat + (size_t)b * NTOK * Dm + d;
  float s = 0.0f;
  for (int n = 0; n < NTOK; n++) s += (float)kb[(size_t)n * Dm];
  KsumEps[idx] = s + 1e-6f;
}

// -------- Z[row] = Q[row,:] . KsumEps[b,:]  (one wave32 per row) ----------
__global__ void z_kernel(const __bf16* __restrict__ Q,
                         const float* __restrict__ KsumEps,
                         float* __restrict__ Z) {
  int row  = blockIdx.x * 8 + (threadIdx.x >> 5);
  int lane = threadIdx.x & 31;
  int b    = row >> 11;   // row / NTOK
  const __bf16* q  = Q + (size_t)row * Dm;
  const float*  ks = KsumEps + (size_t)b * Dm;
  float s = 0.0f;
  for (int d = lane; d < Dm; d += 32) s += (float)q[d] * ks[d];
  for (int off = 16; off > 0; off >>= 1) s += __shfl_xor(s, off, 32);
  if (lane == 0) Z[row] = s;
}

// -------- pooled = x[:, -1, :] (exact f32 copy) ---------------------------
__global__ void pooled_copy(const float* __restrict__ X, float* __restrict__ out) {
  int t = blockIdx.x * blockDim.x + threadIdx.x;   // b*Dm + d
  int b = t >> 10, d = t & (Dm - 1);
  out[t] = X[((size_t)b * NTOK + (NTOK - 1)) * Dm + d];
}

extern "C" void kernel_launch(void* const* d_in, const int* in_sizes, int n_in,
                              void* d_out, int out_size, void* d_ws, size_t ws_size,
                              hipStream_t stream) {
  const int*   ids = (const int*)d_in[0];
  const float* emb = (const float*)d_in[1];
  const float* Wq  = (const float*)d_in[2];
  const float* bq  = (const float*)d_in[3];
  const float* Wk  = (const float*)d_in[4];
  const float* bk  = (const float*)d_in[5];
  const float* Wv  = (const float*)d_in[6];
  const float* bv  = (const float*)d_in[7];
  const float* Wo  = (const float*)d_in[8];
  const float* bo  = (const float*)d_in[9];
  const float* Wh  = (const float*)d_in[10];
  const float* bh  = (const float*)d_in[11];
  float* out = (float*)d_out;

  // ---- workspace carve-up (regions stay 256B-aligned) ----
  char* w = (char*)d_ws;
  float* Xf  = (float*)w;               w += (size_t)MROW * Dm * 4;      // 16 MB
  float* KsE = (float*)w;               w += (size_t)NB * Dm * 4;
  float* Zb  = (float*)w;               w += (size_t)NB * NTOK * 4;
  __bf16* Xb  = (__bf16*)w;             w += (size_t)MROW * Dm * 2;      // 8 MB
  __bf16* Qb  = (__bf16*)w;             w += (size_t)MROW * Dm * 2;
  __bf16* Kb  = (__bf16*)w;             w += (size_t)MROW * Dm * 2;
  __bf16* Vb  = (__bf16*)w;             w += (size_t)MROW * Dm * 2;
  __bf16* CTb = (__bf16*)w;             w += (size_t)MROW * Dm * 2;
  __bf16* KVb = (__bf16*)w;             w += (size_t)NB * Dm * Dm * 2;   // 4 MB
  __bf16* Wbf = (__bf16*)w;             w += (size_t)Dm * Dm * 2;        // 2 MB (reused)
  __bf16* Whb = (__bf16*)w;             /* 50257*1024*2 ~ 103 MB, fits L2 */

  const long NW  = (long)Dm * Dm;            // weight elems
  const int  CVW = (int)(NW / 8 / 256);      // cvt grid for one weight

  gather_embed<<<MROW, 256, 0, stream>>>(ids, emb, Xf, Xb);

  dim3 gFull(MROW / 256, Dm / 64, 1);
  for (int l = 0; l < 2; l++) {
    const float* bql = bq + (size_t)l * Dm;
    const float* bkl = bk + (size_t)l * Dm;
    const float* bvl = bv + (size_t)l * Dm;
    const float* bol = bo + (size_t)l * Dm;

    // Q = elu(X Wq^T + bq)+1
    cvt_f32_bf16<<<CVW, 256, 0, stream>>>(Wq + (size_t)l * NW, Wbf, NW);
    gemm_wmma<<<gFull, 256, 0, stream>>>(Xb, Wbf, bql, nullptr, nullptr, Qb,
                                         Dm, Dm, Dm, 0, 0, 0, 0, 1);
    // K = elu(X Wk^T + bk)+1
    cvt_f32_bf16<<<CVW, 256, 0, stream>>>(Wk + (size_t)l * NW, Wbf, NW);
    gemm_wmma<<<gFull, 256, 0, stream>>>(Xb, Wbf, bkl, nullptr, nullptr, Kb,
                                         Dm, Dm, Dm, 0, 0, 0, 0, 1);
    // V = X Wv^T + bv
    cvt_f32_bf16<<<CVW, 256, 0, stream>>>(Wv + (size_t)l * NW, Wbf, NW);
    gemm_wmma<<<gFull, 256, 0, stream>>>(Xb, Wbf, bvl, nullptr, nullptr, Vb,
                                         Dm, Dm, Dm, 0, 0, 0, 0, 0);

    // KVt[b] = (K^T V)^T per batch
    kv_wmma<<<dim3(Dm / 128, Dm / 64, NB), 256, 0, stream>>>(Kb, Vb, KVb);

    // Ksum + eps, then Z = Q . KsumEps
    ksum_kernel<<<(NB * Dm) / 256, 256, 0, stream>>>(Kb, KsE);
    z_kernel<<<MROW / 8, 256, 0, stream>>>(Qb, KsE, Zb);

    // context = (Q @ KV) / Z   (per-batch B operand, per-row divisor)
    gemm_wmma<<<dim3(NTOK / 256, Dm / 64, NB), 256, 0, stream>>>(
        Qb, KVb, nullptr, Zb, nullptr, CTb, Dm, Dm, Dm,
        (long)NTOK * Dm, (long)Dm * Dm, (long)NTOK * Dm, (long)NTOK, 0);

    // x = context @ Wo^T + bo  (write f32 for pooled + bf16 for next GEMMs)
    cvt_f32_bf16<<<CVW, 256, 0, stream>>>(Wo + (size_t)l * NW, Wbf, NW);
    gemm_wmma<<<gFull, 256, 0, stream>>>(CTb, Wbf, bol, nullptr, Xf, Xb,
                                         Dm, Dm, Dm, 0, 0, 0, 0, 0);
  }

  // head: convert W_head once (103 MB bf16 -> resident in 192 MB L2),
  // then logits = x @ W_head^T + b_head
  const long NH = (long)VOC * Dm;
  cvt_f32_bf16<<<(int)(NH / 8 / 256), 256, 0, stream>>>(Wh, Whb, NH);
  gemm_wmma<<<dim3(MROW / 256, (VOC + 63) / 64, 1), 256, 0, stream>>>(
      Xb, Whb, bh, nullptr, out, nullptr, Dm, VOC, VOC, 0, 0, 0, 0, 0);

  // pooled = x[:, -1]
  pooled_copy<<<(NB * Dm) / 256, 256, 0, stream>>>(Xf, out + (size_t)MROW * VOC);
}
